// CrossAttention_27427661152291
// MI455X (gfx1250) — compile-verified, matrix operand order checked by reading, not run on hardware
//
#include <hip/hip_runtime.h>

typedef __attribute__((ext_vector_type(16))) __bf16 v16bf;
typedef __attribute__((ext_vector_type(8)))  __bf16 v8bf;
typedef __attribute__((ext_vector_type(8)))  float  v8f;

#define CC 256
#define HH 192
#define WW 192
#define HW (HH*WW)   // 36864
#define NB 8

__device__ __forceinline__ v16bf cat16(v8bf a, v8bf b) {
  return __builtin_shufflevector(a, b, 0,1,2,3,4,5,6,7,8,9,10,11,12,13,14,15);
}

// one-time fp32 -> bf16 conversion (for the 256x256 weight matrices)
// grid*block*8 == n
__global__ __launch_bounds__(256) void cvt_f32_bf16_kernel(
    const float* __restrict__ src, __bf16* __restrict__ dst)
{
  const int i = (blockIdx.x * 256 + threadIdx.x) * 8;
  float4 a = *(const float4*)(src + i);
  float4 b = *(const float4*)(src + i + 4);
  v8bf r;
  r[0] = (__bf16)a.x; r[1] = (__bf16)a.y; r[2] = (__bf16)a.z; r[3] = (__bf16)a.w;
  r[4] = (__bf16)b.x; r[5] = (__bf16)b.y; r[6] = (__bf16)b.z; r[7] = (__bf16)b.w;
  *(v8bf*)(dst + i) = r;
}

// y[b,o,n] = sum_i Wb[o,i] * x[b,i,n] + bias[o]   (Wb bf16, output bf16)
// grid.x = B*HW/16 ; block = 256 threads (8 waves); each WG: 16 spatial x 256 out-ch
__global__ __launch_bounds__(256) void conv1x1_bf16_kernel(
    const float* __restrict__ x, const __bf16* __restrict__ Wb,
    const float* __restrict__ bias, __bf16* __restrict__ y)
{
  constexpr int XTS = 264;              // padded bf16 row stride (conflict-free b128)
  __shared__ __bf16 XT[16 * XTS];       // XT[n][i] : transposed input tile

  const int nt    = blockIdx.x;
  const int b     = nt / (HW / 16);
  const int nbase = (nt % (HW / 16)) * 16;
  const int tid   = threadIdx.x;

  { // stage: thread t loads x[b, i=t, nbase..nbase+15], stores transposed as bf16
    const float* src = x + (size_t)b * CC * HW + (size_t)tid * HW + nbase;
    float4 f0 = *(const float4*)(src + 0);
    float4 f1 = *(const float4*)(src + 4);
    float4 f2 = *(const float4*)(src + 8);
    float4 f3 = *(const float4*)(src + 12);
    float vv[16] = {f0.x,f0.y,f0.z,f0.w, f1.x,f1.y,f1.z,f1.w,
                    f2.x,f2.y,f2.z,f2.w, f3.x,f3.y,f3.z,f3.w};
    #pragma unroll
    for (int n = 0; n < 16; ++n) XT[n * XTS + tid] = (__bf16)vv[n];
  }
  __syncthreads();

  const int wv = tid >> 5, lane = tid & 31;
  const int lr = lane & 15, hf = lane >> 4;

  const int o0 = wv * 32 + lr;          // A-operand row for acc0
  const int o1 = o0 + 16;               // A-operand row for acc1
  const __bf16* w0   = Wb + o0 * CC;
  const __bf16* w1   = Wb + o1 * CC;
  const __bf16* brow = &XT[lr * XTS];   // B-operand column n = lr

  v8f acc0 = {0,0,0,0,0,0,0,0};
  v8f acc1 = {0,0,0,0,0,0,0,0};

  #pragma unroll
  for (int kk = 0; kk < CC; kk += 32) {
    v16bf A0 = cat16(*(const v8bf*)(w0 + kk + hf*8),
                     *(const v8bf*)(w0 + kk + 16 + hf*8));
    v16bf A1 = cat16(*(const v8bf*)(w1 + kk + hf*8),
                     *(const v8bf*)(w1 + kk + 16 + hf*8));
    v16bf Bm = cat16(*(const v8bf*)(brow + kk + hf*8),
                     *(const v8bf*)(brow + kk + 16 + hf*8));
    acc0 = __builtin_amdgcn_wmma_f32_16x16x32_bf16(false, A0, false, Bm, (short)0, acc0, false, false);
    acc1 = __builtin_amdgcn_wmma_f32_16x16x32_bf16(false, A1, false, Bm, (short)0, acc1, false, false);
  }

  const size_t outb = (size_t)b * CC * HW + nbase + lr;
  #pragma unroll
  for (int j = 0; j < 8; ++j) {
    int oa = wv * 32 + j + 8 * hf;      // C/D layout: M = j + 8*half
    y[outb + (size_t)oa * HW] = (__bf16)(acc0[j] + bias[oa]);
    int ob = oa + 16;
    y[outb + (size_t)ob * HW] = (__bf16)(acc1[j] + bias[ob]);
  }
}

// One WG per (b,c): S = Q K^T (LDS, fp32) -> softmax -> O = P V
// grid.x = B*C = 2048 ; block = 256 threads (8 waves)
__global__ __launch_bounds__(256) void attn_bf16_kernel(
    const __bf16* __restrict__ qb, const __bf16* __restrict__ kb,
    const __bf16* __restrict__ vb, float* __restrict__ out)
{
  constexpr int SS = 196;               // S row stride (floats), conflict-free b128
  constexpr int VS = 200;               // VT row stride (bf16), conflict-free b128
  __shared__ float  Sbuf[HH * SS];      // 150528 B (P bf16 written in-place later)
  __shared__ __bf16 VT[WW * VS];        // 76800 B : VT[w][g] = v[g][w]

  const size_t base = (size_t)blockIdx.x * HW;
  const __bf16* q = qb + base;
  const __bf16* k = kb + base;
  const __bf16* v = vb + base;
  float*        o = out + base;

  const int tid = threadIdx.x;
  const int wv = tid >> 5, lane = tid & 31;
  const int lr = lane & 15, hf = lane >> 4;

  // transpose V into LDS (192*24 chunks of 8, exactly 18 per thread -> no divergence)
  for (int idx = tid; idx < HH * (WW / 8); idx += 256) {
    int g  = idx / (WW / 8);
    int wc = (idx % (WW / 8)) * 8;
    v8bf d = *(const v8bf*)(v + g * WW + wc);
    #pragma unroll
    for (int j = 0; j < 8; ++j) VT[(wc + j) * VS + g] = d[j];
  }

  // phase 1: S tiles (144 tiles, 18 per wave), operands straight from global
  for (int t = wv; t < 144; t += 8) {
    int ht = t / 12, gt = t % 12;
    const __bf16* qr = q + (ht * 16 + lr) * WW;   // A row h, contiguous along w
    const __bf16* kr = k + (gt * 16 + lr) * WW;   // B col g, contiguous along w
    v8f acc = {0,0,0,0,0,0,0,0};
    #pragma unroll 3
    for (int kk = 0; kk < WW; kk += 32) {
      v16bf A  = cat16(*(const v8bf*)(qr + kk + hf*8), *(const v8bf*)(qr + kk + 16 + hf*8));
      v16bf Bm = cat16(*(const v8bf*)(kr + kk + hf*8), *(const v8bf*)(kr + kk + 16 + hf*8));
      acc = __builtin_amdgcn_wmma_f32_16x16x32_bf16(false, A, false, Bm, (short)0, acc, false, false);
    }
    int col = gt * 16 + lr;
    #pragma unroll
    for (int j = 0; j < 8; ++j) Sbuf[(ht * 16 + j + 8 * hf) * SS + col] = acc[j];
  }
  __syncthreads();

  // phase 2: softmax, one row per thread; write P bf16 in-place (2g <= 4g: safe)
  if (tid < HH) {
    float* row = &Sbuf[tid * SS];
    float m = row[0];
    for (int g = 1; g < HH; ++g) m = fmaxf(m, row[g]);
    float s = 0.f;
    for (int g = 0; g < HH; ++g) { float e = __expf(row[g] - m); row[g] = e; s += e; }
    float inv = 1.0f / s;
    __bf16* pr = (__bf16*)row;
    for (int g = 0; g < HH; ++g) pr[g] = (__bf16)(row[g] * inv);
  }
  __syncthreads();

  // phase 3: O tiles, P rows + VT rows from LDS (both contiguous b128 chunks)
  for (int t = wv; t < 144; t += 8) {
    int ht = t / 12, wt = t % 12;
    const __bf16* pr = (const __bf16*)&Sbuf[(ht * 16 + lr) * SS];
    const __bf16* vr = &VT[(wt * 16 + lr) * VS];
    v8f acc = {0,0,0,0,0,0,0,0};
    #pragma unroll 3
    for (int kk = 0; kk < HH; kk += 32) {
      v16bf A  = cat16(*(const v8bf*)(pr + kk + hf*8), *(const v8bf*)(pr + kk + 16 + hf*8));
      v16bf Bm = cat16(*(const v8bf*)(vr + kk + hf*8), *(const v8bf*)(vr + kk + 16 + hf*8));
      acc = __builtin_amdgcn_wmma_f32_16x16x32_bf16(false, A, false, Bm, (short)0, acc, false, false);
    }
    #pragma unroll
    for (int j = 0; j < 8; ++j)
      o[(ht * 16 + j + 8 * hf) * WW + wt * 16 + lr] = acc[j];
  }
}

extern "C" void kernel_launch(void* const* d_in, const int* in_sizes, int n_in,
                              void* d_out, int out_size, void* d_ws, size_t ws_size,
                              hipStream_t stream) {
  const float* query = (const float*)d_in[0];
  const float* key_  = (const float*)d_in[1];
  const float* value = (const float*)d_in[2];
  const float* Wq = (const float*)d_in[3]; const float* bq = (const float*)d_in[4];
  const float* Wk = (const float*)d_in[5]; const float* bk = (const float*)d_in[6];
  const float* Wv = (const float*)d_in[7]; const float* bv = (const float*)d_in[8];
  float* out = (float*)d_out;

  const size_t N  = (size_t)NB * CC * HW;  // elements per activation tensor
  const size_t NW = (size_t)CC * CC;       // elements per weight matrix
  // workspace: 3 bf16 activation tensors (452,984,832 B) + 3 bf16 weights (393,216 B)
  __bf16* qws = reinterpret_cast<__bf16*>(d_ws);
  __bf16* kws = qws + N;
  __bf16* vws = kws + N;
  __bf16* wqb = vws + N;
  __bf16* wkb = wqb + NW;
  __bf16* wvb = wkb + NW;

  const int cvtGrid = (int)(NW / (256 * 8));  // 32 blocks per matrix
  cvt_f32_bf16_kernel<<<cvtGrid, 256, 0, stream>>>(Wq, wqb);
  cvt_f32_bf16_kernel<<<cvtGrid, 256, 0, stream>>>(Wk, wkb);
  cvt_f32_bf16_kernel<<<cvtGrid, 256, 0, stream>>>(Wv, wvb);

  const int convGrid = NB * HW / 16;          // 18432
  conv1x1_bf16_kernel<<<convGrid, 256, 0, stream>>>(query, wqb, bq, qws);
  conv1x1_bf16_kernel<<<convGrid, 256, 0, stream>>>(key_,  wkb, bk, kws);
  conv1x1_bf16_kernel<<<convGrid, 256, 0, stream>>>(value, wvb, bv, vws);

  attn_bf16_kernel<<<NB * CC, 256, 0, stream>>>(qws, kws, vws, out);
}
